// MANA_46591805227133
// MI455X (gfx1250) — matrix operand order repaired and checked by reading, not compiled, not run on hardware
//
#include <hip/hip_runtime.h>
#include <hip/hip_bf16.h>
#include <math.h>

#define N_ATOMS 50000
#define N_EDGES 400000
#define N_MOL   500
#define HID     128
#define H3      (3 * HID)
#define NRBF    20
#define KPAD    32
#define NLAYERS 4
#define CUTOFF  5.0f

typedef __attribute__((ext_vector_type(16))) __bf16 v16bf;
typedef __attribute__((ext_vector_type(8)))  float  v8f;

static __device__ __forceinline__ unsigned short f2bf_bits(float f) {
    unsigned int u = __float_as_uint(f);
    u += 0x7FFFu + ((u >> 16) & 1u);            // round-to-nearest-even
    return (unsigned short)(u >> 16);
}
static __device__ __forceinline__ float bf2f(unsigned short b) {
    return __uint_as_float(((unsigned int)b) << 16);
}
static __device__ __forceinline__ __bf16 bfbits(unsigned short b) {
    union { unsigned short u; __bf16 h; } c; c.u = b; return c.h;
}
static __device__ __forceinline__ float silu_f(float x) {
    return x / (1.0f + __expf(-x));
}

// ---------------------------------------------------------------------------
// Weight shuffle: (Ksrc x N) f32 row-major -> fragment-major bf16, K padded.
// dst element i (per layer) = ((kt*(N/16)+nt)*32 + lane)*16 + e holds
// W[kt*32 + (lane>>4)*16 + e][nt*16 + (lane&15)], zero-filled for k >= Ksrc.
// A lane's 16 B-fragment values are then CONTIGUOUS (2 x b128 loads).
// ---------------------------------------------------------------------------
__global__ void shuffle_w(const float* __restrict__ src, unsigned short* __restrict__ dst,
                          int K, int N, int Ksrc, int L) {
    int i = blockIdx.x * blockDim.x + threadIdx.x;
    int per = K * N;
    if (i >= L * per) return;
    int l = i / per;
    int r = i % per;
    int e    = r & 15;
    int lane = (r >> 4) & 31;
    int tile = r >> 9;
    int ntiles = N >> 4;
    int nt = tile % ntiles;
    int kt = tile / ntiles;
    int k = kt * 32 + (lane >> 4) * 16 + e;
    int n = nt * 16 + (lane & 15);
    float v = (k < Ksrc) ? src[(size_t)l * Ksrc * N + (size_t)k * N + n] : 0.f;
    dst[i] = f2bf_bits(v);
}

// ---------------------------------------------------------------------------
// WMMA GEMM: C(MxN) = act(A(MxK,bf16 row-major) * Bsh(fragment-major) + bias)
// One wave computes a 16x64 tile (4 accumulators, A-fragment reused 4x).
// 8 waves (256 threads) per block -> 128x64 block tile.
// Requires M%16==0 per tile (guarded), N%64==0, K%32==0.
// ---------------------------------------------------------------------------
template <int ACT, int OUTBF>
__global__ void wmma_gemm(const unsigned short* __restrict__ A,
                          const unsigned short* __restrict__ Bsh,
                          const float* __restrict__ bias,
                          float* __restrict__ Cf,
                          unsigned short* __restrict__ Cb,
                          int M, int N, int K) {
    const int lane = threadIdx.x & 31;
    const int wave = threadIdx.x >> 5;
    const int tM   = (blockIdx.y * 8 + wave) * 16;
    const int tN   = blockIdx.x * 64;
    if (tM >= M) return;

    const int m      = lane & 15;   // row within A-frag / col within B,C-frag
    const int hgrp   = lane >> 4;   // lane half selector
    const int ntiles = N >> 4;

    v8f acc[4];
#pragma unroll
    for (int j = 0; j < 4; ++j) acc[j] = (v8f){0.f,0.f,0.f,0.f,0.f,0.f,0.f,0.f};

    const unsigned short* Arow = A + (size_t)(tM + m) * K;

    for (int k0 = 0; k0 < K; k0 += 32) {
        // A fragment (16x32 bf16): one contiguous 16-element run k = 16*half + i -> 2 x b128
        v16bf a;
#pragma unroll
        for (int i = 0; i < 8; ++i) a[i]     = bfbits(Arow[k0 + hgrp * 16 + i]);
#pragma unroll
        for (int i = 0; i < 8; ++i) a[8 + i] = bfbits(Arow[k0 + hgrp * 16 + 8 + i]);

        const int kt = k0 >> 5;
        const unsigned short* Bbase =
            Bsh + ((size_t)(kt * ntiles + (tN >> 4)) * 32 + lane) * 16;
#pragma unroll
        for (int j = 0; j < 4; ++j) {
            // fragment-major: 16 contiguous bf16 per lane -> 2 x b128
            const unsigned short* Bp = Bbase + (size_t)j * 32 * 16;
            v16bf b;
#pragma unroll
            for (int i = 0; i < 16; ++i) b[i] = bfbits(Bp[i]);
            acc[j] = __builtin_amdgcn_wmma_f32_16x16x32_bf16(
                false, a, false, b, (short)0, acc[j], false, false);
        }
    }

#pragma unroll
    for (int j = 0; j < 4; ++j) {
        const int nn = tN + j * 16 + m;
        const float bs = bias ? bias[nn] : 0.f;
#pragma unroll
        for (int r = 0; r < 8; ++r) {
            const int mm = tM + hgrp * 8 + r;   // C/D layout: M = half*8 + r
            float val = acc[j][r] + bs;
            if (ACT) val = silu_f(val);
            const size_t idx = (size_t)mm * N + nn;
            if (OUTBF) Cb[idx] = f2bf_bits(val);
            else       Cf[idx] = val;
        }
    }
}

// ---------------------------------------------------------------------------
// Edge geometry: dist, dirs, bf16 RBF (padded K=20 -> 32 with zeros)
// ---------------------------------------------------------------------------
__global__ void edge_geom(const float* __restrict__ pos,
                          const int* __restrict__ row, const int* __restrict__ col,
                          float* __restrict__ dirs, unsigned short* __restrict__ rbfb) {
    int e = blockIdx.x * blockDim.x + threadIdx.x;
    if (e >= N_EDGES) return;
    int r = row[e], c = col[e];
    float dx = pos[c * 3 + 0] - pos[r * 3 + 0];
    float dy = pos[c * 3 + 1] - pos[r * 3 + 1];
    float dz = pos[c * 3 + 2] - pos[r * 3 + 2];
    float d   = sqrtf(dx * dx + dy * dy + dz * dz);
    float inv = 1.0f / (d + 1e-8f);
    dirs[e * 3 + 0] = dx * inv;
    dirs[e * 3 + 1] = dy * inv;
    dirs[e * 3 + 2] = dz * inv;
    const float step = CUTOFF / (float)(NRBF - 1);
    unsigned short* rb = rbfb + (size_t)e * KPAD;
#pragma unroll
    for (int j = 0; j < NRBF; ++j) {
        float t = d - (float)j * step;
        rb[j] = f2bf_bits(__expf(-t * t));
    }
#pragma unroll
    for (int j = NRBF; j < KPAD; ++j) rb[j] = 0;
}

__global__ void init_s(const int* __restrict__ z, const float* __restrict__ emb,
                       float* __restrict__ s) {
    int i = blockIdx.x * blockDim.x + threadIdx.x;
    if (i >= N_ATOMS * HID) return;
    int n = i / HID, h = i % HID;
    s[i] = emb[(size_t)z[n] * HID + h];
}

// ---------------------------------------------------------------------------
// Per-(edge,channel) message construction + atomic scatter to nodes
// ---------------------------------------------------------------------------
__global__ void message_scatter(const unsigned short* __restrict__ Hb,
                                const float* __restrict__ s, const float* __restrict__ v,
                                const int* __restrict__ row, const int* __restrict__ col,
                                const float* __restrict__ dirs,
                                float* __restrict__ m_s, float* __restrict__ m_v) {
    int i = blockIdx.x * blockDim.x + threadIdx.x;
    if (i >= N_EDGES * HID) return;
    int e = i / HID, h = i % HID;
    int r = row[e], c = col[e];
    const unsigned short* he = Hb + (size_t)e * H3;
    float pss = bf2f(he[h]);
    float pvv = bf2f(he[HID + h]);
    float psv = bf2f(he[2 * HID + h]);
    float sc  = s[(size_t)c * HID + h];
    atomicAdd(&m_s[(size_t)r * HID + h], pss * sc);
    float svd = psv * sc;
    size_t vc = ((size_t)c * HID + h) * 3;
    size_t vr = ((size_t)r * HID + h) * 3;
#pragma unroll
    for (int k = 0; k < 3; ++k)
        atomicAdd(&m_v[vr + k], pvv * v[vc + k] + svd * dirs[e * 3 + k]);
}

// ---------------------------------------------------------------------------
// v_norm + build bf16 concat [s | m_s | ||m_v||] for the update MLP
// ---------------------------------------------------------------------------
__global__ void build_x(const float* __restrict__ s, const float* __restrict__ m_s,
                        const float* __restrict__ m_v, unsigned short* __restrict__ Xb) {
    int i = blockIdx.x * blockDim.x + threadIdx.x;
    if (i >= N_ATOMS * HID) return;
    int n = i / HID, h = i % HID;
    size_t b = (size_t)i * 3;
    float x = m_v[b], y = m_v[b + 1], zc = m_v[b + 2];
    float nv = sqrtf(x * x + y * y + zc * zc);
    unsigned short* X = Xb + (size_t)n * H3;
    X[h]           = f2bf_bits(s[i]);
    X[HID + h]     = f2bf_bits(m_s[i]);
    X[2 * HID + h] = f2bf_bits(nv);
}

__global__ void update_sv(const float* __restrict__ u, const float* __restrict__ m_v,
                          float* __restrict__ s, float* __restrict__ v) {
    int i = blockIdx.x * blockDim.x + threadIdx.x;
    if (i >= N_ATOMS * HID) return;
    int n = i / HID, h = i % HID;
    const float* un = u + (size_t)n * H3;
    s[i] += un[h];
    float alpha = un[HID + h], beta = un[2 * HID + h];
    size_t b = (size_t)i * 3;
#pragma unroll
    for (int k = 0; k < 3; ++k) v[b + k] = alpha * v[b + k] + beta * m_v[b + k];
}

__global__ void pool_mol(const float* __restrict__ s, const int* __restrict__ batch,
                         float* __restrict__ h_mol, float* __restrict__ counts) {
    int i = blockIdx.x * blockDim.x + threadIdx.x;
    if (i >= N_ATOMS * HID) return;
    int n = i / HID, h = i % HID;
    atomicAdd(&h_mol[(size_t)batch[n] * HID + h], s[i]);
    if (h == 0) atomicAdd(&counts[batch[n]], 1.0f);
}

// One block per molecule (500 blocks x 128 threads) — tiny, scalar path.
__global__ void readout(const float* __restrict__ h_mol, const float* __restrict__ counts,
                        const float* __restrict__ lW1, const float* __restrict__ lb1,
                        const float* __restrict__ lW2, const float* __restrict__ lb2,
                        const float* __restrict__ pW1, const float* __restrict__ pb1,
                        const float* __restrict__ pW2, const float* __restrict__ pb2,
                        float* __restrict__ out) {
    __shared__ float hm[HID];
    __shared__ float red[HID];
    int m = blockIdx.x, t = threadIdx.x;
    float cnt = counts[m];
    if (cnt < 1.0f) cnt = 1.0f;
    hm[t] = h_mol[(size_t)m * HID + t] / cnt;
    __syncthreads();

    float acc = lb1[t];
    for (int k = 0; k < HID; ++k) acc += hm[k] * lW1[k * HID + t];
    red[t] = silu_f(acc) * lW2[t];
    __syncthreads();
    for (int sN = 64; sN > 0; sN >>= 1) {
        if (t < sN) red[t] += red[t + sN];
        __syncthreads();
    }
    if (t == 0) out[m] = red[0] + lb2[0];
    __syncthreads();

    float acc2 = pb1[t];
    for (int k = 0; k < HID; ++k) acc2 += hm[k] * pW1[k * HID + t];
    red[t] = silu_f(acc2) * pW2[t];
    __syncthreads();
    for (int sN = 64; sN > 0; sN >>= 1) {
        if (t < sN) red[t] += red[t + sN];
        __syncthreads();
    }
    if (t == 0) {
        float x = red[0] + pb2[0];
        out[N_MOL + m] = 1.0f / (1.0f + __expf(-x));
    }
}

// ---------------------------------------------------------------------------
extern "C" void kernel_launch(void* const* d_in, const int* in_sizes, int n_in,
                              void* d_out, int out_size, void* d_ws, size_t ws_size,
                              hipStream_t stream) {
    (void)in_sizes; (void)n_in; (void)out_size; (void)ws_size;
    const int*   z    = (const int*)d_in[0];
    const float* pos  = (const float*)d_in[1];
    const int*   edge = (const int*)d_in[2];
    const int*   row  = edge;
    const int*   col  = edge + N_EDGES;
    const int*   batch = (const int*)d_in[3];
    const float* emb  = (const float*)d_in[4];
    const float* fW1  = (const float*)d_in[5];
    const float* fb1  = (const float*)d_in[6];
    const float* fW2  = (const float*)d_in[7];
    const float* fb2  = (const float*)d_in[8];
    const float* uW1  = (const float*)d_in[9];
    const float* ub1  = (const float*)d_in[10];
    const float* uW2  = (const float*)d_in[11];
    const float* ub2  = (const float*)d_in[12];
    const float* lW1  = (const float*)d_in[13];
    const float* lb1  = (const float*)d_in[14];
    const float* lW2  = (const float*)d_in[15];
    const float* lb2  = (const float*)d_in[16];
    const float* pW1  = (const float*)d_in[17];
    const float* pb1  = (const float*)d_in[18];
    const float* pW2  = (const float*)d_in[19];
    const float* pb2  = (const float*)d_in[20];
    float* out = (float*)d_out;

    // ---- workspace carve-up -------------------------------------------------
    char* ws = (char*)d_ws;
    size_t off = 0;
    auto carve = [&](size_t bytes) -> char* {
        char* p = ws + off;
        off += (bytes + 255) & ~(size_t)255;
        return p;
    };
    float*          dirs  = (float*)carve((size_t)N_EDGES * 3 * 4);
    unsigned short* rbfb  = (unsigned short*)carve((size_t)N_EDGES * KPAD * 2);
    unsigned short* h1b   = (unsigned short*)carve((size_t)N_EDGES * HID * 2);
    unsigned short* Hb    = (unsigned short*)carve((size_t)N_EDGES * H3 * 2);
    float*          sbuf  = (float*)carve((size_t)N_ATOMS * HID * 4);
    float*          vbuf  = (float*)carve((size_t)N_ATOMS * HID * 3 * 4);
    float*          m_s   = (float*)carve((size_t)N_ATOMS * HID * 4);
    float*          m_v   = (float*)carve((size_t)N_ATOMS * HID * 3 * 4);
    unsigned short* Xb    = (unsigned short*)carve((size_t)N_ATOMS * H3 * 2);
    unsigned short* hidb  = (unsigned short*)carve((size_t)N_ATOMS * HID * 2);
    float*          ubuf  = (float*)carve((size_t)N_ATOMS * H3 * 4);
    unsigned short* fW1b  = (unsigned short*)carve((size_t)NLAYERS * KPAD * HID * 2);
    unsigned short* fW2b  = (unsigned short*)carve((size_t)NLAYERS * HID * H3 * 2);
    unsigned short* uW1b  = (unsigned short*)carve((size_t)NLAYERS * H3 * HID * 2);
    unsigned short* uW2b  = (unsigned short*)carve((size_t)NLAYERS * HID * H3 * 2);
    float*          h_mol = (float*)carve((size_t)N_MOL * HID * 4);
    float*          cnts  = (float*)carve((size_t)N_MOL * 4);

    const int NH  = N_ATOMS * HID;   // 6.4M
    const int EH  = N_EDGES * HID;   // 51.2M
    const int TPB = 256;

    // ---- weights -> fragment-major bf16 ------------------------------------
    {
        int n1 = NLAYERS * KPAD * HID;
        shuffle_w<<<(n1 + TPB - 1) / TPB, TPB, 0, stream>>>(fW1, fW1b, KPAD, HID, NRBF, NLAYERS);
        int n2 = NLAYERS * HID * H3;
        shuffle_w<<<(n2 + TPB - 1) / TPB, TPB, 0, stream>>>(fW2, fW2b, HID, H3, HID, NLAYERS);
        int n3 = NLAYERS * H3 * HID;
        shuffle_w<<<(n3 + TPB - 1) / TPB, TPB, 0, stream>>>(uW1, uW1b, H3, HID, H3, NLAYERS);
        shuffle_w<<<(n2 + TPB - 1) / TPB, TPB, 0, stream>>>(uW2, uW2b, HID, H3, HID, NLAYERS);
    }

    // ---- geometry + state init ---------------------------------------------
    edge_geom<<<(N_EDGES + TPB - 1) / TPB, TPB, 0, stream>>>(pos, row, col, dirs, rbfb);
    init_s<<<(NH + TPB - 1) / TPB, TPB, 0, stream>>>(z, emb, sbuf);
    hipMemsetAsync(vbuf, 0, (size_t)N_ATOMS * HID * 3 * 4, stream);

    const dim3 blk(256);                                   // 8 waves
    const dim3 gE128(HID / 64, (N_EDGES / 16 + 7) / 8);    // (2, 3125)
    const dim3 gE384(H3 / 64, (N_EDGES / 16 + 7) / 8);     // (6, 3125)
    const dim3 gN128(HID / 64, (N_ATOMS / 16 + 7) / 8);    // (2, 391)
    const dim3 gN384(H3 / 64, (N_ATOMS / 16 + 7) / 8);     // (6, 391)

    for (int l = 0; l < NLAYERS; ++l) {
        hipMemsetAsync(m_s, 0, (size_t)N_ATOMS * HID * 4, stream);
        hipMemsetAsync(m_v, 0, (size_t)N_ATOMS * HID * 3 * 4, stream);

        // edge filter MLP: silu(rbf @ W1 + b1) @ W2 + b2
        wmma_gemm<1, 1><<<gE128, blk, 0, stream>>>(
            rbfb, fW1b + (size_t)l * KPAD * HID, fb1 + (size_t)l * HID,
            nullptr, h1b, N_EDGES, HID, KPAD);
        wmma_gemm<0, 1><<<gE384, blk, 0, stream>>>(
            h1b, fW2b + (size_t)l * HID * H3, fb2 + (size_t)l * H3,
            nullptr, Hb, N_EDGES, H3, HID);

        // message construction + segment-sum scatter
        message_scatter<<<(EH + TPB - 1) / TPB, TPB, 0, stream>>>(
            Hb, sbuf, vbuf, row, col, dirs, m_s, m_v);

        // node update MLP
        build_x<<<(NH + TPB - 1) / TPB, TPB, 0, stream>>>(sbuf, m_s, m_v, Xb);
        wmma_gemm<1, 1><<<gN128, blk, 0, stream>>>(
            Xb, uW1b + (size_t)l * H3 * HID, ub1 + (size_t)l * HID,
            nullptr, hidb, N_ATOMS, HID, H3);
        wmma_gemm<0, 0><<<gN384, blk, 0, stream>>>(
            hidb, uW2b + (size_t)l * HID * H3, ub2 + (size_t)l * H3,
            ubuf, nullptr, N_ATOMS, H3, HID);
        update_sv<<<(NH + TPB - 1) / TPB, TPB, 0, stream>>>(ubuf, m_v, sbuf, vbuf);
    }

    // ---- readout ------------------------------------------------------------
    hipMemsetAsync(h_mol, 0, (size_t)N_MOL * HID * 4, stream);
    hipMemsetAsync(cnts, 0, (size_t)N_MOL * 4, stream);
    pool_mol<<<(NH + TPB - 1) / TPB, TPB, 0, stream>>>(sbuf, batch, h_mol, cnts);
    readout<<<N_MOL, HID, 0, stream>>>(h_mol, cnts, lW1, lb1, lW2, lb2,
                                       pW1, pb1, pW2, pb2, out);
}
